// DiscreteAttention_48335561949594
// MI455X (gfx1250) — matrix operand order repaired and checked by reading, not compiled, not run on hardware
//
#include <hip/hip_runtime.h>
#include <stdint.h>

// ---------------- problem constants (from reference) ----------------
#define B_        8
#define N_        16384
#define D_        256
#define K_        16
#define NITER_    5
#define EPS_      1e-8f
#define TAU_      1.0f

// em_iter tiling: 8 waves/block, each wave handles GROUPS groups of 32 rows.
// GROUPS=1 -> 512 workgroups (B*N/256): ~2 per WGP on a ~256-WGP part, giving
// enough waves in flight to saturate HBM/L2 for this streaming-bound kernel.
#define WAVES_    8
#define GROUPS_   1
#define ROWS_PER_BLOCK (WAVES_ * GROUPS_ * 32)        // 256
#define BLOCKS_PER_BATCH (N_ / ROWS_PER_BLOCK)        // 64

typedef __attribute__((ext_vector_type(16))) __bf16 v16bf;
typedef __attribute__((ext_vector_type(8)))  float  v8f;

union Frag16 { v16bf v; uint4 q[2]; unsigned short h[16]; };
union FragC  { v8f v; float f[8]; };

__device__ __forceinline__ uint16_t f2bf(float f) {
  uint32_t u = __float_as_uint(f);
  u += 0x7fffu + ((u >> 16) & 1u);           // round-to-nearest-even
  return (uint16_t)(u >> 16);
}

// deterministic hash -> Gumbel(0,1)  (stand-in for jax.random.gumbel(key(1)))
__device__ __forceinline__ float gumbel_hash(uint32_t idx) {
  uint32_t h = idx * 747796405u + 2891336453u;
  uint32_t x = ((h >> ((h >> 28u) + 4u)) ^ h) * 277803737u;
  x = (x >> 22u) ^ x;
  float u = ((float)x + 0.5f) * (1.0f / 4294967296.0f);
  return -__logf(-__logf(u));
}

// ------------------------------------------------------------------
// prep: p = softmax(inputs) -> bf16 row-major; x -> bf16 transposed
// (xT[b][d][n]) via LDS tile transpose. One block = 16 rows of one batch.
// ------------------------------------------------------------------
__global__ void __launch_bounds__(256) prep_kernel(
    const float* __restrict__ in, uint16_t* __restrict__ p16,
    uint16_t* __restrict__ xT) {
  __shared__ __align__(16) uint16_t sx[16 * D_];
  const int b    = blockIdx.x / (N_ / 16);
  const int t16  = blockIdx.x % (N_ / 16);
  const int lane = threadIdx.x & 31;
  const int w    = threadIdx.x >> 5;

  #pragma unroll
  for (int rr = 0; rr < 2; ++rr) {
    const int r = w * 2 + rr;                       // row within 16-row tile
    const long long row = (long long)b * N_ + t16 * 16 + r;
    const float* src = in + row * D_;
    float x[8];
    float m = -INFINITY;
    #pragma unroll
    for (int i = 0; i < 8; ++i) { x[i] = src[lane + 32 * i]; m = fmaxf(m, x[i]); }
    #pragma unroll
    for (int msk = 16; msk >= 1; msk >>= 1) m = fmaxf(m, __shfl_xor(m, msk, 32));
    float s = 0.f;
    #pragma unroll
    for (int i = 0; i < 8; ++i) { x[i] = __expf(x[i] - m); s += x[i]; }
    #pragma unroll
    for (int msk = 16; msk >= 1; msk >>= 1) s += __shfl_xor(s, msk, 32);
    const float inv = 1.f / s;
    uint16_t* pd = p16 + row * D_;
    #pragma unroll
    for (int i = 0; i < 8; ++i) {
      const int d = lane + 32 * i;
      pd[d]            = f2bf(x[i] * inv);
      sx[r * D_ + d]   = f2bf(src[d]);
    }
  }
  __syncthreads();
  // transpose: thread t owns d = t, writes 16 contiguous n values (32B)
  const int d = threadIdx.x;
  __align__(16) uint16_t vals[16];
  #pragma unroll
  for (int r = 0; r < 16; ++r) vals[r] = sx[r * D_ + d];
  uint16_t* dst = xT + ((long long)b * D_ + d) * N_ + t16 * 16;
  *(uint4*)(dst)     = *(const uint4*)(vals);
  *(uint4*)(dst + 8) = *(const uint4*)(vals + 8);
}

// ------------------------------------------------------------------
// init: log_q = log_softmax(slot_logits) broadcast over B (bf16);
//       log_pi = log(mix + eps). 16 waves, one per slot k.
// ------------------------------------------------------------------
__global__ void __launch_bounds__(512) init_kernel(
    const float* __restrict__ slot_logits, const float* __restrict__ mix,
    uint16_t* __restrict__ logq, float* __restrict__ logpi) {
  const int lane = threadIdx.x & 31;
  const int k    = threadIdx.x >> 5;                 // 0..15
  const float* src = slot_logits + k * D_;
  float o[8];
  float m = -INFINITY;
  #pragma unroll
  for (int i = 0; i < 8; ++i) { o[i] = src[lane + 32 * i]; m = fmaxf(m, o[i]); }
  #pragma unroll
  for (int msk = 16; msk >= 1; msk >>= 1) m = fmaxf(m, __shfl_xor(m, msk, 32));
  float s = 0.f;
  #pragma unroll
  for (int i = 0; i < 8; ++i) s += __expf(o[i] - m);
  #pragma unroll
  for (int msk = 16; msk >= 1; msk >>= 1) s += __shfl_xor(s, msk, 32);
  const float lse = m + __logf(s);
  for (int b = 0; b < B_; ++b) {
    uint16_t* q = logq + (b * K_ + k) * D_;
    #pragma unroll
    for (int i = 0; i < 8; ++i) q[lane + 32 * i] = f2bf(o[i] - lse);
  }
  if (threadIdx.x < B_ * K_)
    logpi[threadIdx.x] = __logf(mix[threadIdx.x & (K_ - 1)] + EPS_);
}

__global__ void zero_kernel(float* __restrict__ numer, float* __restrict__ S) {
  const int i = blockIdx.x * 256 + threadIdx.x;
  if (i < B_ * K_ * D_) numer[i] = 0.f;
  if (i < B_ * K_)      S[i]     = 0.f;
}

// ------------------------------------------------------------------
// em_iter: fused  C[k][n] = log_q . p^T   (WMMA bf16, A=log_q, B=p)
//                 gamma   = softmax_k(C + log_pi)  (k lives IN-LANE: 8 + 1 shfl)
//                 S      += sum_n gamma  ;  numer += gamma^T . x  (WMMA bf16)
// ------------------------------------------------------------------
__global__ void __launch_bounds__(256, 1) em_iter_kernel(
    const uint16_t* __restrict__ p16, const uint16_t* __restrict__ xT,
    const uint16_t* __restrict__ logq, const float* __restrict__ logpi,
    float* __restrict__ numer, float* __restrict__ Sg) {
  __shared__ __align__(16) uint16_t s_logq[K_ * D_];      // 8 KB
  __shared__ float    s_logpi[K_];
  __shared__ float    s_S[K_];
  __shared__ __align__(16) float    s_num[K_ * D_];       // 16 KB
  __shared__ __align__(16) uint16_t s_gT[WAVES_][K_ * 32];// 8 KB, per-wave gamma^T

  const int tid  = threadIdx.x;
  const int lane = tid & 31;
  const int w    = tid >> 5;
  const int b    = blockIdx.x / BLOCKS_PER_BATCH;
  const int blk  = blockIdx.x % BLOCKS_PER_BATCH;
  const int row0blk = blk * ROWS_PER_BLOCK;

  // stage log_q (bf16) and log_pi into LDS
  *(uint4*)(&s_logq[tid * 16]) = *(const uint4*)(&logq[(size_t)b * K_ * D_ + tid * 16]);
  if (tid < K_) { s_logpi[tid] = logpi[b * K_ + tid]; s_S[tid] = 0.f; }
  for (int i = tid; i < K_ * D_; i += 256) s_num[i] = 0.f;
  __syncthreads();

  const int kcol = lane & 15;       // WMMA 16-wide index (row for A, col for B/C)
  const int half = lane >> 4;       // 0 or 1

  // preload A-fragments of log_q (M=k, Kdim=d), reused for every row tile.
  // A layout: lane row k = l%16; halves 0-7: d = 32c+8h+{0..7}; 8-15: +16.
  Frag16 aq[8];
  #pragma unroll
  for (int c = 0; c < 8; ++c) {
    const uint16_t* sp = &s_logq[kcol * D_ + 32 * c + 8 * half];
    aq[c].q[0] = *(const uint4*)sp;
    aq[c].q[1] = *(const uint4*)(sp + 16);
  }

  // per-lane log_pi for the 8 in-lane k values (k = j + 8*half)
  float lp[8];
  #pragma unroll
  for (int j = 0; j < 8; ++j) lp[j] = s_logpi[j + 8 * half];

  FragC acc[16];                    // numerator partials: 16 d-chunks
  #pragma unroll
  for (int dc = 0; dc < 16; ++dc)
    #pragma unroll
    for (int j = 0; j < 8; ++j) acc[dc].f[j] = 0.f;

  float sacc[8];                    // per-lane S[k=j+8h] partials
  #pragma unroll
  for (int j = 0; j < 8; ++j) sacc[j] = 0.f;

  for (int g = 0; g < GROUPS_; ++g) {
    const int n0 = row0blk + (w * GROUPS_ + g) * 32;
    #pragma unroll
    for (int s = 0; s < 2; ++s) {                       // two 16-row subtiles
      const int r0 = n0 + s * 16;
      FragC cfr;
      #pragma unroll
      for (int j = 0; j < 8; ++j) cfr.f[j] = 0.f;
      // B = p tile: lane col n = l%16 -> row r0+kcol; halves 0-7: d = 32c+8h+{0..7}; 8-15: +16
      const long long prow = ((long long)b * N_ + r0 + kcol) * D_;
      #pragma unroll
      for (int c = 0; c < 8; ++c) {                     // GEMM1 over D in 32-chunks
        Frag16 bp;
        const uint16_t* ap = p16 + prow + 32 * c + 8 * half;
        bp.q[0] = *(const uint4*)ap;
        bp.q[1] = *(const uint4*)(ap + 16);
        cfr.v = __builtin_amdgcn_wmma_f32_16x16x32_bf16(
            false, aq[c].v, false, bp.v, (short)0, cfr.v, false, false);
      }
      // softmax over k: 8 in-lane values + partner half (one shfl each)
      float v[8];
      float m = -INFINITY;
      #pragma unroll
      for (int j = 0; j < 8; ++j) { v[j] = cfr.f[j] + lp[j]; m = fmaxf(m, v[j]); }
      m = fmaxf(m, __shfl_xor(m, 16, 32));
      float ssum = 0.f;
      #pragma unroll
      for (int j = 0; j < 8; ++j) { v[j] = __expf(v[j] - m); ssum += v[j]; }
      ssum += __shfl_xor(ssum, 16, 32);
      const float inv = 1.f / ssum;
      #pragma unroll
      for (int j = 0; j < 8; ++j) {
        const float gm = v[j] * inv;                    // gamma[n=r0+kcol][k=j+8h]
        sacc[j] += gm;
        s_gT[w][(j + 8 * half) * 32 + s * 16 + kcol] = f2bf(gm);
      }
    }
    // per-wave LDS RAW (cross-lane): wait for DS stores before fragment reads
    asm volatile("s_wait_dscnt 0x0" ::: "memory");
    Frag16 ag;                                           // A = gamma^T (16k x 32n)
    {
      const uint16_t* gp = &s_gT[w][kcol * 32 + 8 * half];
      ag.q[0] = *(const uint4*)gp;
      ag.q[1] = *(const uint4*)(gp + 16);
    }
    #pragma unroll
    for (int dc = 0; dc < 16; ++dc) {                    // GEMM2 over D in 16-col tiles
      Frag16 bx;
      const uint16_t* xp = xT + ((long long)b * D_ + dc * 16 + kcol) * N_ + n0 + 8 * half;
      bx.q[0] = *(const uint4*)xp;
      bx.q[1] = *(const uint4*)(xp + 16);
      acc[dc].v = __builtin_amdgcn_wmma_f32_16x16x32_bf16(
          false, ag.v, false, bx.v, (short)0, acc[dc].v, false, false);
    }
  }

  // flush per-lane S partials once per block
  #pragma unroll
  for (int j = 0; j < 8; ++j) atomicAdd(&s_S[j + 8 * half], sacc[j]);

  // reduce numerator partials through LDS, then one global atomic pass
  #pragma unroll
  for (int dc = 0; dc < 16; ++dc)
    #pragma unroll
    for (int j = 0; j < 8; ++j)
      atomicAdd(&s_num[(j + 8 * half) * D_ + dc * 16 + kcol], acc[dc].f[j]);
  __syncthreads();
  for (int i = tid; i < K_ * D_; i += 256)
    atomicAdd(&numer[(size_t)b * K_ * D_ + i], s_num[i]);
  if (tid < K_) atomicAdd(&Sg[b * K_ + tid], s_S[tid]);
}

// ------------------------------------------------------------------
// update: theta = numer / (S+eps); logq = log_softmax(theta) (bf16);
//         logpi = log(S/N + eps). One wave per (b,k).
// ------------------------------------------------------------------
__global__ void __launch_bounds__(32) update_kernel(
    const float* __restrict__ numer, const float* __restrict__ Sg,
    float* __restrict__ theta, uint16_t* __restrict__ logq,
    float* __restrict__ logpi) {
  const int bk = blockIdx.x, lane = threadIdx.x;
  const float S = Sg[bk];
  const float inv = 1.f / (S + EPS_);
  const float* nrow = numer + (size_t)bk * D_;
  float t[8];
  float m = -INFINITY;
  #pragma unroll
  for (int i = 0; i < 8; ++i) { t[i] = nrow[lane + 32 * i] * inv; m = fmaxf(m, t[i]); }
  #pragma unroll
  for (int msk = 16; msk >= 1; msk >>= 1) m = fmaxf(m, __shfl_xor(m, msk, 32));
  float s = 0.f;
  #pragma unroll
  for (int i = 0; i < 8; ++i) s += __expf(t[i] - m);
  #pragma unroll
  for (int msk = 16; msk >= 1; msk >>= 1) s += __shfl_xor(s, msk, 32);
  const float lse = m + __logf(s);
  float*    trow = theta + (size_t)bk * D_;
  uint16_t* qrow = logq + (size_t)bk * D_;
  #pragma unroll
  for (int i = 0; i < 8; ++i) {
    const int d = lane + 32 * i;
    trow[d] = t[i];
    qrow[d] = f2bf(t[i] - lse);
  }
  if (lane == 0) logpi[bk] = __logf(S * (1.f / N_) + EPS_);
}

// ------------------------------------------------------------------
// final: slots = softmax((theta + gumbel)/tau). One wave per (b,k).
// ------------------------------------------------------------------
__global__ void __launch_bounds__(32) final_kernel(
    const float* __restrict__ theta, float* __restrict__ out) {
  const int bk = blockIdx.x, lane = threadIdx.x;
  const float* trow = theta + (size_t)bk * D_;
  float v[8];
  float m = -INFINITY;
  #pragma unroll
  for (int i = 0; i < 8; ++i) {
    const int d = lane + 32 * i;
    v[i] = (trow[d] + gumbel_hash((uint32_t)(bk * D_ + d))) * (1.f / TAU_);
    m = fmaxf(m, v[i]);
  }
  #pragma unroll
  for (int msk = 16; msk >= 1; msk >>= 1) m = fmaxf(m, __shfl_xor(m, msk, 32));
  float s = 0.f;
  #pragma unroll
  for (int i = 0; i < 8; ++i) { v[i] = __expf(v[i] - m); s += v[i]; }
  #pragma unroll
  for (int msk = 16; msk >= 1; msk >>= 1) s += __shfl_xor(s, msk, 32);
  const float inv = 1.f / s;
  #pragma unroll
  for (int i = 0; i < 8; ++i) out[(size_t)bk * D_ + lane + 32 * i] = v[i] * inv;
}

// ------------------------------------------------------------------
extern "C" void kernel_launch(void* const* d_in, const int* in_sizes, int n_in,
                              void* d_out, int out_size, void* d_ws, size_t ws_size,
                              hipStream_t stream) {
  (void)in_sizes; (void)n_in; (void)out_size; (void)ws_size;
  const float* inputs = (const float*)d_in[0];          // (B,N,256) f32
  const float* slot   = (const float*)d_in[1];          // (1,16,256) f32
  const float* mix    = (const float*)d_in[2];          // (1,16) f32

  char* w = (char*)d_ws;
  const size_t szBND = (size_t)B_ * N_ * D_ * sizeof(uint16_t);   // 64 MiB
  uint16_t* p16   = (uint16_t*)w;                 w += szBND;
  uint16_t* xT16  = (uint16_t*)w;                 w += szBND;
  uint16_t* logq  = (uint16_t*)w;                 w += (size_t)B_ * K_ * D_ * 2;
  float*    logpi = (float*)w;                    w += (size_t)B_ * K_ * 4;
  float*    Sacc  = (float*)w;                    w += (size_t)B_ * K_ * 4;
  float*    numer = (float*)w;                    w += (size_t)B_ * K_ * D_ * 4;
  float*    theta = (float*)w;

  prep_kernel<<<B_ * N_ / 16, 256, 0, stream>>>(inputs, p16, xT16);
  init_kernel<<<1, 512, 0, stream>>>(slot, mix, logq, logpi);
  for (int it = 0; it < NITER_; ++it) {
    zero_kernel<<<(B_ * K_ * D_ + 255) / 256, 256, 0, stream>>>(numer, Sacc);
    em_iter_kernel<<<B_ * BLOCKS_PER_BATCH, 256, 0, stream>>>(
        p16, xT16, logq, logpi, numer, Sacc);
    update_kernel<<<B_ * K_, 32, 0, stream>>>(numer, Sacc, theta, logq, logpi);
  }
  final_kernel<<<B_ * K_, 32, 0, stream>>>(theta, (float*)d_out);
}
